// RelativeTimeAttention_70162585748101
// MI455X (gfx1250) — compile-verified, matrix operand order checked by reading, not run on hardware
//
#include <hip/hip_runtime.h>
#include <math.h>

typedef unsigned short u16;
typedef __attribute__((ext_vector_type(16))) __bf16 v16bf;
typedef __attribute__((ext_vector_type(8)))  float  v8f;

union Frag  { v16bf v; uint4 q[2]; };
union Pack8 { uint4 q; u16 s[8]; };

#define BATCH  4
#define SEQ    2048
#define DMODEL 1024
#define NHEAD  16
#define HDIM   64

__device__ __forceinline__ u16 f2bf(float f) {
    unsigned u = __builtin_bit_cast(unsigned, f);
    u += 0x7fffu + ((u >> 16) & 1u);          // round-to-nearest-even
    return (u16)(u >> 16);
}

// async copy 16B global -> LDS (no VGPR data path, tracked by ASYNCcnt)
__device__ __forceinline__ void async_b128(const u16* gptr, u16* lptr) {
    unsigned laddr = (unsigned)(unsigned long long)lptr;   // generic addr low 32b == LDS offset
    asm volatile("global_load_async_to_lds_b128 %0, %1, off"
                 :: "v"(laddr), "v"(gptr) : "memory");
}
__device__ __forceinline__ void wait_async0() {
    asm volatile("s_wait_asynccnt 0x0" ::: "memory");
}

// ---------------- fp32 -> bf16 conversion ----------------
__global__ __launch_bounds__(256) void cvt_f32_bf16(const float* __restrict__ in,
                                                    u16* __restrict__ out, int n4) {
    int i = blockIdx.x * 256 + threadIdx.x;
    if (i >= n4) return;
    float4 f = ((const float4*)in)[i];
    uint2 p;
    p.x = (unsigned)f2bf(f.x) | ((unsigned)f2bf(f.y) << 16);
    p.y = (unsigned)f2bf(f.z) | ((unsigned)f2bf(f.w) << 16);
    ((uint2*)out)[i] = p;
}

// fp32 W[K][N] -> bf16 Wt[N][K]  (so GEMM B-fragments load row-major [n][k])
__global__ __launch_bounds__(256) void cvt_transpose(const float* __restrict__ W,
                                                     u16* __restrict__ Wt) {
    int k = blockIdx.x * 16 + threadIdx.x;
    int n = blockIdx.y * 16 + threadIdx.y;
    Wt[(size_t)n * DMODEL + k] = f2bf(W[(size_t)k * DMODEL + n]);
}

// ---------------- tiled bf16 WMMA GEMM: C[M,N] = A[M,K] @ Wt[N,K]^T + bias ----------------
// 128x128 tile per 256-thread block, BK=32, double-buffered async LDS staging.
__global__ __launch_bounds__(256) void gemm_bf16_wmma(
    const u16* __restrict__ A, const u16* __restrict__ Wt,
    const float* __restrict__ bias, u16* __restrict__ outb,
    float* __restrict__ outf, int write_f32)
{
    __shared__ u16 sA[2][128 * 40];   // pad 40 -> conflict-free b128 fragment loads
    __shared__ u16 sB[2][128 * 40];
    const int t = threadIdx.x;
    const int wave = t >> 5, lane = t & 31;
    const int m0 = blockIdx.y * 128, n0 = blockIdx.x * 128;
    const int wr = (wave >> 1) * 32;   // 4x2 wave grid over the 128x128 tile
    const int wc = (wave & 1) * 64;
    const int lhalf = lane >> 4;       // K-interleave group per ISA A-layout
    const int l15 = lane & 15;

    // this thread's two 16B staging chunks (fixed row/col, varying kk)
    const int row0 = t >> 2, c40 = t & 3;              // chunk = t
    const int row1 = (t + 256) >> 2, c41 = t & 3;      // chunk = t + 256

    v8f acc[2][4];
    for (int i = 0; i < 2; ++i) for (int j = 0; j < 4; ++j) acc[i][j] = {};

    auto fill = [&](int buf, int kk) {
        const u16* ga0 = A  + (size_t)(m0 + row0) * DMODEL + kk + c40 * 8;
        const u16* gb0 = Wt + (size_t)(n0 + row0) * DMODEL + kk + c40 * 8;
        const u16* ga1 = A  + (size_t)(m0 + row1) * DMODEL + kk + c41 * 8;
        const u16* gb1 = Wt + (size_t)(n0 + row1) * DMODEL + kk + c41 * 8;
        async_b128(ga0, &sA[buf][row0 * 40 + c40 * 8]);
        async_b128(gb0, &sB[buf][row0 * 40 + c40 * 8]);
        async_b128(ga1, &sA[buf][row1 * 40 + c41 * 8]);
        async_b128(gb1, &sB[buf][row1 * 40 + c41 * 8]);
        if (kk + 64 < DMODEL) {                         // warm L2 two slabs ahead
            __builtin_prefetch(ga0 + 32, 0, 1);
            __builtin_prefetch(gb0 + 32, 0, 1);
        }
    };

    fill(0, 0);
    wait_async0();
    __syncthreads();

    for (int kk = 0; kk < DMODEL; kk += 32) {
        const int cur = (kk >> 5) & 1;
        if (kk + 32 < DMODEL) fill(cur ^ 1, kk + 32);   // DMA next tile under the WMMAs

        Frag af[2], bf[4];
        const int kb = lhalf * 8;
        for (int sm = 0; sm < 2; ++sm) {
            const u16* p = &sA[cur][(wr + sm * 16 + l15) * 40 + kb];
            af[sm].q[0] = *(const uint4*)(p);
            af[sm].q[1] = *(const uint4*)(p + 16);
        }
        for (int sn = 0; sn < 4; ++sn) {
            const u16* p = &sB[cur][(wc + sn * 16 + l15) * 40 + kb];
            bf[sn].q[0] = *(const uint4*)(p);
            bf[sn].q[1] = *(const uint4*)(p + 16);
        }
        for (int sm = 0; sm < 2; ++sm)
            for (int sn = 0; sn < 4; ++sn)
                acc[sm][sn] = __builtin_amdgcn_wmma_f32_16x16x32_bf16(
                    false, af[sm].v, false, bf[sn].v, (short)0, acc[sm][sn], false, false);

        wait_async0();          // next tile landed
        __syncthreads();        // everyone done reading cur; nxt visible to all
    }

    for (int sm = 0; sm < 2; ++sm)
        for (int sn = 0; sn < 4; ++sn)
            for (int r = 0; r < 8; ++r) {
                int m = m0 + wr + sm * 16 + r + lhalf * 8;   // C-layout row mapping
                int n = n0 + wc + sn * 16 + l15;
                float v = acc[sm][sn][r] + bias[n];
                if (write_f32) outf[(size_t)m * DMODEL + n] = v;
                else           outb[(size_t)m * DMODEL + n] = f2bf(v);
            }
}

// ---------------- flash attention with time-decay bias, bf16 WMMA ----------------
__global__ __launch_bounds__(256) void attn_wmma(
    const u16* __restrict__ Q, const u16* __restrict__ K, const u16* __restrict__ V,
    const float* __restrict__ vt, const float* __restrict__ scale,
    const int* __restrict__ lengths, u16* __restrict__ O)
{
    __shared__ u16 sK[2][32 * 72];     // [32 keys][64 d]  (pad 72)
    __shared__ u16 sV[2][64 * 40];     // [64 d][32 keys]  (transposed, pad 40)
    __shared__ u16 sP[8 * 16 * 40];    // per-wave P scratch [16 q][32 keys]

    const int t = threadIdx.x, wave = t >> 5, lane = t & 31;
    const int lhalf = lane >> 4, l15 = lane & 15;
    const int h = blockIdx.y, b = blockIdx.z;
    const int qbase = (blockIdx.x * 8 + wave) * 16;
    const int len = lengths[b];
    const float sc = scale[h];
    const float* vtb = vt + (size_t)b * SEQ;
    const float L2E = 1.4426950408889634f;

    // Q fragments (two 16x32 k-halves of the 16x64 q-tile), loaded straight from global
    Frag aq[2];
    {
        const u16* qrow = Q + (((size_t)(b * SEQ + qbase + l15)) * NHEAD + h) * HDIM;
        for (int kh = 0; kh < 2; ++kh) {
            int db = kh * 32 + lhalf * 8;
            aq[kh].q[0] = *(const uint4*)(qrow + db);
            aq[kh].q[1] = *(const uint4*)(qrow + db + 16);
        }
    }
    float vtq[8];
    for (int r = 0; r < 8; ++r) vtq[r] = vtb[qbase + r + lhalf * 8];

    float mi[8], li[8];
    for (int r = 0; r < 8; ++r) { mi[r] = -1e30f; li[r] = 0.0f; }
    v8f o[4];
    for (int j = 0; j < 4; ++j) o[j] = {};

    u16* sPw = sP + wave * (16 * 40);

    auto stageK = [&](int buf, int kt) {      // async direct copy, row-major
        int row = t >> 3, c = t & 7;
        const u16* src = K + (((size_t)(b * SEQ + kt + row)) * NHEAD + h) * HDIM + c * 8;
        async_b128(src, &sK[buf][row * 72 + c * 8]);
    };
    auto stageV = [&](int buf, int kt) {      // transpose needs the VGPR path
        int key = t & 31, d0 = (t >> 5) * 8;
        Pack8 pv;
        pv.q = *(const uint4*)(V + (((size_t)(b * SEQ + kt + key)) * NHEAD + h) * HDIM + d0);
        for (int j = 0; j < 8; ++j) sV[buf][(d0 + j) * 40 + key] = pv.s[j];
    };

    stageK(0, 0);
    stageV(0, 0);
    wait_async0();
    __syncthreads();

    for (int kt = 0; kt < SEQ; kt += 32) {
        const int cur = (kt >> 5) & 1;
        if (kt + 32 < SEQ) {                  // stage next key tile under this one's math
            stageK(cur ^ 1, kt + 32);
            stageV(cur ^ 1, kt + 32);
        }

        // QK^T: two 16x16 score tiles, each = two K=32 WMMAs over HD=64
        float sv[2][8];
        for (int kn = 0; kn < 2; ++kn) {
            Frag bk[2];
            const u16* kp = &sK[cur][(kn * 16 + l15) * 72];
            for (int kh = 0; kh < 2; ++kh) {
                int db = kh * 32 + lhalf * 8;
                bk[kh].q[0] = *(const uint4*)(kp + db);
                bk[kh].q[1] = *(const uint4*)(kp + db + 16);
            }
            v8f s = {};
            s = __builtin_amdgcn_wmma_f32_16x16x32_bf16(false, aq[0].v, false, bk[0].v, (short)0, s, false, false);
            s = __builtin_amdgcn_wmma_f32_16x16x32_bf16(false, aq[1].v, false, bk[1].v, (short)0, s, false, false);
            int key = kt + kn * 16 + l15;
            float vtk = vtb[key];
            bool dead = (key >= len);
            for (int r = 0; r < 8; ++r) {
                float x = s[r] * 0.125f - fabsf(vtq[r] - vtk) * sc;
                sv[kn][r] = dead ? -1e30f : x;
            }
        }

        // online softmax (row reductions across the 16-lane N groups of the C layout)
        float pm[2][8];
        for (int r = 0; r < 8; ++r) {
            float tm = fmaxf(sv[0][r], sv[1][r]);
            for (int m = 1; m < 16; m <<= 1) tm = fmaxf(tm, __shfl_xor(tm, m, 32));
            float mnew  = fmaxf(mi[r], tm);
            float alpha = exp2f((mi[r] - mnew) * L2E);
            float p0 = exp2f((sv[0][r] - mnew) * L2E);
            float p1 = exp2f((sv[1][r] - mnew) * L2E);
            float rs = p0 + p1;
            for (int m = 1; m < 16; m <<= 1) rs += __shfl_xor(rs, m, 32);
            li[r] = li[r] * alpha + rs;
            mi[r] = mnew;
            pm[0][r] = p0; pm[1][r] = p1;
            for (int j = 0; j < 4; ++j) o[j][r] *= alpha;
        }

        // C-layout -> A-layout for P via wave-private LDS round-trip
        for (int kn = 0; kn < 2; ++kn)
            for (int r = 0; r < 8; ++r)
                sPw[(r + lhalf * 8) * 40 + kn * 16 + l15] = f2bf(pm[kn][r]);
        asm volatile("s_wait_dscnt 0x0" ::: "memory");

        Frag ap;
        {
            const u16* pp = sPw + l15 * 40 + lhalf * 8;
            ap.q[0] = *(const uint4*)(pp);
            ap.q[1] = *(const uint4*)(pp + 16);
        }
        for (int sn = 0; sn < 4; ++sn) {
            Frag bv;
            const u16* vp = &sV[cur][(sn * 16 + l15) * 40 + lhalf * 8];
            bv.q[0] = *(const uint4*)(vp);
            bv.q[1] = *(const uint4*)(vp + 16);
            o[sn] = __builtin_amdgcn_wmma_f32_16x16x32_bf16(
                false, ap.v, false, bv.v, (short)0, o[sn], false, false);
        }

        wait_async0();          // next K tile landed
        __syncthreads();        // V-transpose stores visible; cur safe to overwrite
    }

    for (int sn = 0; sn < 4; ++sn)
        for (int r = 0; r < 8; ++r) {
            int q = qbase + r + lhalf * 8;
            int d = sn * 16 + l15;
            float v = o[sn][r] / li[r];
            O[(((size_t)(b * SEQ + q)) * NHEAD + h) * HDIM + d] = f2bf(v);
        }
}

// ---------------- host-side launch ----------------
extern "C" void kernel_launch(void* const* d_in, const int* in_sizes, int n_in,
                              void* d_out, int out_size, void* d_ws, size_t ws_size,
                              hipStream_t stream) {
    const float* x   = (const float*)d_in[0];
    const float* vt  = (const float*)d_in[1];
    const float* Wq  = (const float*)d_in[2];
    const float* bq  = (const float*)d_in[3];
    const float* Wk  = (const float*)d_in[4];
    const float* bk  = (const float*)d_in[5];
    const float* Wv  = (const float*)d_in[6];
    const float* bv  = (const float*)d_in[7];
    const float* Wo  = (const float*)d_in[8];
    const float* bo  = (const float*)d_in[9];
    const float* scale   = (const float*)d_in[10];
    const int*   lengths = (const int*)d_in[11];
    float* out = (float*)d_out;

    char* ws = (char*)d_ws;
    const size_t MB = 1024ull * 1024ull;
    u16* Xb  = (u16*)(ws + 0);        // 16 MB
    u16* Wqt = (u16*)(ws + 16 * MB);  //  2 MB each
    u16* Wkt = (u16*)(ws + 18 * MB);
    u16* Wvt = (u16*)(ws + 20 * MB);
    u16* Wot = (u16*)(ws + 22 * MB);
    u16* Qb  = (u16*)(ws + 24 * MB);  // 16 MB each
    u16* Kb  = (u16*)(ws + 40 * MB);
    u16* Vb  = (u16*)(ws + 56 * MB);
    u16* Ab  = (u16*)(ws + 72 * MB);  // attention output (bf16)

    const int M = BATCH * SEQ;        // 8192
    int n4 = M * DMODEL / 4;
    cvt_f32_bf16<<<(n4 + 255) / 256, 256, 0, stream>>>(x, Xb, n4);
    dim3 tb(16, 16), tg(DMODEL / 16, DMODEL / 16);
    cvt_transpose<<<tg, tb, 0, stream>>>(Wq, Wqt);
    cvt_transpose<<<tg, tb, 0, stream>>>(Wk, Wkt);
    cvt_transpose<<<tg, tb, 0, stream>>>(Wv, Wvt);
    cvt_transpose<<<tg, tb, 0, stream>>>(Wo, Wot);

    dim3 gg(DMODEL / 128, M / 128);
    gemm_bf16_wmma<<<gg, 256, 0, stream>>>(Xb, Wqt, bq, Qb, nullptr, 0);
    gemm_bf16_wmma<<<gg, 256, 0, stream>>>(Xb, Wkt, bk, Kb, nullptr, 0);
    gemm_bf16_wmma<<<gg, 256, 0, stream>>>(Xb, Wvt, bv, Vb, nullptr, 0);

    dim3 ga(SEQ / 128, NHEAD, BATCH);
    attn_wmma<<<ga, 256, 0, stream>>>(Qb, Kb, Vb, vt, scale, lengths, Ab);

    gemm_bf16_wmma<<<gg, 256, 0, stream>>>(Ab, Wot, bo, nullptr, out, 1);
}
